// SupConLoss_850403524935
// MI455X (gfx1250) — matrix operand order repaired and checked by reading, not compiled, not run on hardware
//
#include <hip/hip_runtime.h>
#include <hip/hip_bf16.h>
#include <math.h>

// ---------------------------------------------------------------------------
// SupConLoss, fused for MI455X (gfx1250, wave32, WMMA):
//   X[N,C] bf16 (N=8192, C=128) -> Gram tiles via v_wmma_f32_16x16x32_bf16,
//   folded on the fly into per-row {sum exp, masked logit sum, mask count}.
//   B tiles are staged into LDS once per workgroup (8 waves share them) with
//   double-buffered async global->LDS copies overlapping the WMMA+exp work.
// ---------------------------------------------------------------------------

typedef __attribute__((ext_vector_type(16))) __bf16 bf16x16;
typedef __attribute__((ext_vector_type(8)))  __bf16 bf16x8;
typedef __attribute__((ext_vector_type(8)))  float  f32x8;
typedef __attribute__((ext_vector_type(4)))  int    i32x4;

typedef __attribute__((address_space(1))) i32x4* gptr_i32x4;
typedef __attribute__((address_space(3))) i32x4* lptr_i32x4;

constexpr int   Bb  = 4, Vv = 2, Cc = 128, Hh = 32, Ww = 32;
constexpr int   Nn  = Bb * Vv * Hh * Ww;        // 8192
constexpr float INV_T = 1.0f / 0.07f;

#if __has_builtin(__builtin_amdgcn_global_load_async_to_lds_b128)
#define HAVE_ASYNC_LDS 1
#else
#define HAVE_ASYNC_LDS 0
#endif

#if __has_builtin(__builtin_amdgcn_s_wait_asynccnt)
#define WAIT_ASYNC() __builtin_amdgcn_s_wait_asynccnt(0)
#else
#define WAIT_ASYNC() asm volatile("s_wait_asynccnt 0" ::: "memory")
#endif

// ---------------------------------------------------------------------------
// Kernel 1: layout change + f32->bf16 convert + label remap + stat zeroing.
//   n = ((v*B + b)*H + h)*W + w  (views stacked along batch, per reference)
//   Xb[n*C + c] = bf16(features[b, v, c, h, w])
// ---------------------------------------------------------------------------
__global__ void supcon_prep(const float* __restrict__ feat,
                            const int*   __restrict__ labels,
                            unsigned short* __restrict__ Xb,
                            int*   __restrict__ lab,
                            float* __restrict__ S1,
                            float* __restrict__ S2,
                            float* __restrict__ CNT)
{
    int t = blockIdx.x * blockDim.x + threadIdx.x;
    if (t >= Nn * Cc) return;
    int n = t >> 7;          // / C
    int c = t & (Cc - 1);
    int w = n & (Ww - 1);
    int h = (n >> 5)  & (Hh - 1);
    int b = (n >> 10) & (Bb - 1);
    int v =  n >> 12;
    size_t src = ((((size_t)b * Vv + v) * Cc + c) * Hh + h) * Ww + w;
    unsigned int u = __float_as_uint(feat[src]);
    u += 0x7FFFu + ((u >> 16) & 1u);            // round-to-nearest-even bf16
    Xb[(size_t)n * Cc + c] = (unsigned short)(u >> 16);
    if (c == 0) {
        size_t ls = (((size_t)b * Vv + v) * Hh + h) * Ww + w;
        lab[n] = labels[ls];
        S1[n] = 0.0f; S2[n] = 0.0f; CNT[n] = 0.0f;
    }
}

// ---------------------------------------------------------------------------
// Stage one 16-row x 128-col bf16 B-tile (4KB) into LDS, 16B per thread.
// ---------------------------------------------------------------------------
__device__ __forceinline__ void stage_tile(const unsigned short* __restrict__ Xb,
                                           unsigned short* dst, int jbase, int t)
{
    const unsigned short* src = Xb + (size_t)(jbase + (t >> 4)) * Cc + (t & 15) * 8;
    unsigned short*       d   = dst + (t >> 4) * Cc + (t & 15) * 8;
#if HAVE_ASYNC_LDS
    __builtin_amdgcn_global_load_async_to_lds_b128(
        (gptr_i32x4)(size_t)src,
        (lptr_i32x4)d, 0, 0);
#else
    *(int4*)d = *(const int4*)src;     // global_load_b128 + ds_store_b128
#endif
}

// ---------------------------------------------------------------------------
// Kernel 2: fused Gram + softmax statistics.
//   grid = (64 row-blocks of 128 rows, 8 column splits of 1024 cols)
//   block = 256 threads = 8 waves; each wave owns a 16-row stripe; the block
//   walks 64 shared j-tiles with double-buffered LDS staging,
//   4x v_wmma_f32_16x16x32_bf16 per tile (K = 128).
// ---------------------------------------------------------------------------
__global__ __launch_bounds__(256, 1) void supcon_main(
    const unsigned short* __restrict__ Xb,
    const int*   __restrict__ lab,
    float* __restrict__ S1,
    float* __restrict__ S2,
    float* __restrict__ CNT)
{
    __shared__ unsigned short Bt[2][16 * Cc];   // 2 x 4KB ping-pong B tiles
    __shared__ int labLds[1024];                // labels of this column split

    const int tid  = threadIdx.x;
    const int lane = tid & 31;
    const int wave = tid >> 5;
    const int half = (lane >> 4) & 1;   // which 16-lane half of the wave
    const int l16  = lane & 15;

    const int ibase  = blockIdx.x * 128 + wave * 16;
    const int jbase0 = blockIdx.y * 1024;

    // ---- one-time stages: labels of the column split + first B tile -------
    *(int4*)&labLds[tid * 4] = *(const int4*)&lab[jbase0 + tid * 4];
    stage_tile(Xb, Bt[0], jbase0, tid);

    // ---- A operand (rows ibase..ibase+15, all K=128), kept in VGPRs -------
    // A uses contiguous k order (same as B): lane<16 holds K {0..15}, lane>=16 holds
    // K {16..31} of row (ibase + l16)  -> two b128 loads per chunk.
    bf16x16 A[4];
    {
        const unsigned short* ap = Xb + (size_t)(ibase + l16) * Cc;
        #pragma unroll
        for (int kc = 0; kc < 4; ++kc) {
            const int k0 = kc * 32 + half * 16;
            union { bf16x16 v; bf16x8 h[2]; } u;
            u.h[0] = *(const bf16x8*)(ap + k0);
            u.h[1] = *(const bf16x8*)(ap + k0 + 8);
            A[kc] = u.v;
        }
    }

    // Labels for the 8 rows this lane-half contributes to (vgpr r <-> row).
    int labi[8];
    #pragma unroll
    for (int r = 0; r < 8; ++r) labi[r] = lab[ibase + r + half * 8];

    float s1[8], s2[8], cn[8];
    #pragma unroll
    for (int r = 0; r < 8; ++r) { s1[r] = 0.0f; s2[r] = 0.0f; cn[r] = 0.0f; }

    for (int jt = 0; jt < 64; ++jt) {
        const int cur = jt & 1;
#if HAVE_ASYNC_LDS
        WAIT_ASYNC();                       // my async stage of tile jt landed
#endif
        __syncthreads();                    // everyone's tile jt visible; all
                                            // waves done reading buffer cur^1
        if (jt + 1 < 64)                    // overlap next copy with compute
            stage_tile(Xb, Bt[cur ^ 1], jbase0 + (jt + 1) * 16, tid);

        // B operand 32x16: B[k][n] = X[jbase+n][k]; lane holds 16 consecutive
        // K values of row (jbase + l16): 32 contiguous bytes from LDS.
        const unsigned short* bp = &Bt[cur][l16 * Cc + half * 16];
        f32x8 acc = {};
        #pragma unroll
        for (int kc = 0; kc < 4; ++kc) {
            bf16x16 Bop = *(const bf16x16*)(bp + kc * 32);   // ds_load_b128 x2
            acc = __builtin_amdgcn_wmma_f32_16x16x32_bf16(
                false, A[kc], false, Bop, (short)0, acc, false, false);
        }

        const int jbase = jbase0 + jt * 16;
        const int jcol  = jbase + l16;
        const int labj  = labLds[jt * 16 + l16];
        #pragma unroll
        for (int r = 0; r < 8; ++r) {
            const int   irow = ibase + r + half * 8;
            const float lgt  = acc[r] * INV_T;
            const bool  diag = (irow == jcol);
            s1[r] += diag ? 0.0f : __expf(lgt);       // Sum_j!=i exp(l_ij)
            const float m = (!diag && labi[r] == labj) ? 1.0f : 0.0f;
            s2[r] += m * lgt;                         // Sum_j mask * l_ij
            cn[r] += m;                               // Sum_j mask
        }
    }

    // Each 16-lane half-wave holds one full row per accumulator slot:
    // butterfly-reduce across 16 lanes, then one atomicAdd per row.
    #pragma unroll
    for (int r = 0; r < 8; ++r) {
        float a = s1[r], b = s2[r], c = cn[r];
        #pragma unroll
        for (int m = 8; m >= 1; m >>= 1) {
            a += __shfl_xor(a, m, 16);
            b += __shfl_xor(b, m, 16);
            c += __shfl_xor(c, m, 16);
        }
        if (l16 == 0) {
            const int irow = ibase + r + half * 8;
            atomicAdd(&S1[irow],  a);
            atomicAdd(&S2[irow],  b);
            atomicAdd(&CNT[irow], c);
        }
    }
}

// ---------------------------------------------------------------------------
// Kernel 3: loss_i = log(S1_i) - S2_i/cnt_i; mean over non-background rows.
// ---------------------------------------------------------------------------
__global__ void supcon_final(const float* __restrict__ S1,
                             const float* __restrict__ S2,
                             const float* __restrict__ CNT,
                             const int*   __restrict__ lab,
                             float* __restrict__ out)
{
    __shared__ float redL[256];
    __shared__ float redN[256];
    float sl = 0.0f, sn = 0.0f;
    for (int i = threadIdx.x; i < Nn; i += blockDim.x) {
        const float nb   = (lab[i] != 0) ? 1.0f : 0.0f;
        const float loss = logf(S1[i]) - S2[i] / CNT[i];
        sl += nb * loss;
        sn += nb;
    }
    redL[threadIdx.x] = sl;
    redN[threadIdx.x] = sn;
    __syncthreads();
    for (int s = 128; s > 0; s >>= 1) {
        if ((int)threadIdx.x < s) {
            redL[threadIdx.x] += redL[threadIdx.x + s];
            redN[threadIdx.x] += redN[threadIdx.x + s];
        }
        __syncthreads();
    }
    if (threadIdx.x == 0) out[0] = redL[0] / redN[0];
}

// ---------------------------------------------------------------------------
extern "C" void kernel_launch(void* const* d_in, const int* in_sizes, int n_in,
                              void* d_out, int out_size, void* d_ws, size_t ws_size,
                              hipStream_t stream)
{
    (void)in_sizes; (void)n_in; (void)out_size; (void)ws_size;
    const float* feat   = (const float*)d_in[0];
    const int*   labels = (const int*)d_in[1];

    char* ws = (char*)d_ws;
    unsigned short* Xb = (unsigned short*)ws;                 // N*C bf16 = 2 MB
    size_t off = (size_t)Nn * Cc * sizeof(unsigned short);
    int*   lab = (int*)(ws + off);   off += (size_t)Nn * sizeof(int);
    float* S1  = (float*)(ws + off); off += (size_t)Nn * sizeof(float);
    float* S2  = (float*)(ws + off); off += (size_t)Nn * sizeof(float);
    float* CNT = (float*)(ws + off); off += (size_t)Nn * sizeof(float);

    hipLaunchKernelGGL(supcon_prep, dim3((Nn * Cc + 255) / 256), dim3(256), 0,
                       stream, feat, labels, Xb, lab, S1, S2, CNT);
    hipLaunchKernelGGL(supcon_main, dim3(64, 8), dim3(256), 0,
                       stream, Xb, lab, S1, S2, CNT);
    hipLaunchKernelGGL(supcon_final, dim3(1), dim3(256), 0,
                       stream, S1, S2, CNT, lab, (float*)d_out);
}
